// FieldDualAttentionWrapper_26869315404115
// MI455X (gfx1250) — matrix operand order repaired and checked by reading, not compiled, not run on hardware
//
#include <hip/hip_runtime.h>
#include <hip/hip_bf16.h>

// ---------------------------------------------------------------------------
// Problem constants (from the reference): B=64, S=512, D=H=F=1024
// ---------------------------------------------------------------------------
#define BB     64
#define SS     512
#define DD     1024
#define HH     1024
#define MS     (BB * SS)     // 32768 rows
#define EPSV   1e-6f

#define M_TILE 64            // rows of (b,s) per workgroup in the WMMA kernel
#define PITCH  1032          // LDS row pitch in bf16 elems (1024 + 8 pad -> bank spread)

typedef __attribute__((ext_vector_type(16))) __bf16        v16bf;
typedef __attribute__((ext_vector_type(8)))  float         v8f;
typedef __attribute__((ext_vector_type(8)))  unsigned int  v8u;
typedef __attribute__((ext_vector_type(4)))  unsigned int  u32x4;

__device__ __forceinline__ unsigned short f32_to_bf16_rne(float x) {
    unsigned int u = __builtin_bit_cast(unsigned int, x);
    unsigned int r = u + 0x7FFFu + ((u >> 16) & 1u);   // round-to-nearest-even
    return (unsigned short)(r >> 16);
}

__device__ __forceinline__ v16bf make_frag(u32x4 lo, u32x4 hi) {
    v8u r;
    r[0] = lo[0]; r[1] = lo[1]; r[2] = lo[2]; r[3] = lo[3];
    r[4] = hi[0]; r[5] = hi[1]; r[6] = hi[2]; r[7] = hi[3];
    return __builtin_bit_cast(v16bf, r);
}

// branch-free tanh: (e^2x - 1)/(e^2x + 1), clamped so e^2x stays finite.
__device__ __forceinline__ float fast_tanh(float x) {
    float xc = fminf(fmaxf(x, -10.f), 10.f);
    float e  = __expf(2.f * xc);
    return (e - 1.f) / (e + 1.f);
}

// ---------------------------------------------------------------------------
// Kernel 0: convert W1 / W4 (f32, [H,D]) to bf16 in workspace
// ---------------------------------------------------------------------------
__global__ __launch_bounds__(256)
void convert_w_kernel(const float* __restrict__ W1, const float* __restrict__ W4,
                      unsigned short* __restrict__ W1b, unsigned short* __restrict__ W4b) {
    int i = blockIdx.x * 256 + threadIdx.x;          // grid covers H*D = 1M
    W1b[i] = f32_to_bf16_rne(W1[i]);
    W4b[i] = f32_to_bf16_rne(W4[i]);
}

// ---------------------------------------------------------------------------
// Kernel 1: gates  gamma = tanh(x W2^T + b2),  alpha = tanh(x W5^T + b5)
// grid = (H/256, B, 2), block = 256
// ---------------------------------------------------------------------------
__global__ __launch_bounds__(256)
void gate_kernel(const float* __restrict__ inp,
                 const float* __restrict__ W2, const float* __restrict__ b2,
                 const float* __restrict__ W5, const float* __restrict__ b5,
                 float* __restrict__ gamma, float* __restrict__ alpha) {
    __shared__ float xin[DD];
    const int b = blockIdx.y;
    const int h = blockIdx.x * 256 + threadIdx.x;
    const int which = blockIdx.z;
    for (int i = threadIdx.x; i < DD; i += 256) xin[i] = inp[b * DD + i];
    __syncthreads();
    const float* W  = which ? W5 : W2;
    const float* bv = which ? b5 : b2;
    float acc = bv[h];
    const float4* wr = (const float4*)(W + (size_t)h * DD);
#pragma unroll 4
    for (int k4 = 0; k4 < DD / 4; ++k4) {
        float4 w = wr[k4];
        const float* xp = &xin[k4 * 4];
        acc += w.x * xp[0] + w.y * xp[1] + w.z * xp[2] + w.w * xp[3];
    }
    (which ? alpha : gamma)[b * HH + h] = fast_tanh(acc);
}

// ---------------------------------------------------------------------------
// Kernel 2 (the big one): fused GEMM + tanh + gated reduction.
//   logits[m] = sum_h tanh( (A W^T)[m,h] + bias[h] ) * gate[b(m), h]
// A = [B*S, D] f32, W = bf16 [H, D], gate = [B, H].
// 256 threads = 8 waves. WG owns 64 rows staged as bf16 in LDS.
// wave = (mgroup 0..3) x (ngroup 0..1): mgroup picks 16 rows, ngroup picks
// half of the 64 N-tiles. Per K-step one A fragment feeds 4 WMMAs (4 N-tiles),
// with software-pipelined fragment loads (prefetch K+32 while computing K).
// Each ngroup writes a partial logit buffer; softmax kernel sums both.
// ---------------------------------------------------------------------------
__global__ __launch_bounds__(256)
void attn_logits_kernel(const float* __restrict__ Aglob,
                        const unsigned short* __restrict__ Wbf,
                        const float* __restrict__ bias,
                        const float* __restrict__ gate,
                        float* __restrict__ logits_part) {   // [2][B*S]
    extern __shared__ unsigned short Asm[];            // M_TILE * PITCH bf16
    const int tid   = threadIdx.x;
    const int mbase = blockIdx.x * M_TILE;

    // ---- stage A tile (64 x 1024 f32 -> bf16) into LDS, coalesced float4 ----
    for (int idx = tid; idx < M_TILE * (DD / 4); idx += 256) {
        const int r  = idx >> 8;                       // /256
        const int c4 = idx & 255;
        const float4 v = *(const float4*)(Aglob + (size_t)(mbase + r) * DD + c4 * 4);
        unsigned short* p = &Asm[r * PITCH + c4 * 4];
        p[0] = f32_to_bf16_rne(v.x); p[1] = f32_to_bf16_rne(v.y);
        p[2] = f32_to_bf16_rne(v.z); p[3] = f32_to_bf16_rne(v.w);
    }
    __syncthreads();

    const int wave   = tid >> 5;                       // 0..7
    const int lane   = tid & 31;
    const int half   = lane >> 4;                      // 0 | 1
    const int l16    = lane & 15;
    const int mgroup = wave & 3;                       // 16-row group
    const int ngroup = wave >> 2;                      // half of N range
    const int mrow0  = mgroup * 16;
    const int b      = mbase / SS;                     // same batch for whole tile

    // A frag per-lane base: M = l16 (both halves), elems 0..15 contiguous at +kc
    const unsigned short* arow = &Asm[(mrow0 + l16) * PITCH + half * 16];

    float acc[8] = {0.f, 0.f, 0.f, 0.f, 0.f, 0.f, 0.f, 0.f};

    const int nt0 = ngroup * 32;
    for (int ntb = nt0; ntb < nt0 + 32; ntb += 4) {
        // B frag rows: N = l16 for each of the 4 tiles; K elems 0..15 contiguous
        const unsigned short* wr0 = Wbf + (size_t)((ntb + 0) * 16 + l16) * DD + half * 16;
        const unsigned short* wr1 = Wbf + (size_t)((ntb + 1) * 16 + l16) * DD + half * 16;
        const unsigned short* wr2 = Wbf + (size_t)((ntb + 2) * 16 + l16) * DD + half * 16;
        const unsigned short* wr3 = Wbf + (size_t)((ntb + 3) * 16 + l16) * DD + half * 16;

        v8f c0 = {}, c1 = {}, c2 = {}, c3 = {};

        // prolog: fragment raw loads for kc = 0
        u32x4 al  = *(const u32x4*)(arow);
        u32x4 ah  = *(const u32x4*)(arow + 8);
        u32x4 b0l = ((const u32x4*)wr0)[0], b0h = ((const u32x4*)wr0)[1];
        u32x4 b1l = ((const u32x4*)wr1)[0], b1h = ((const u32x4*)wr1)[1];
        u32x4 b2l = ((const u32x4*)wr2)[0], b2h = ((const u32x4*)wr2)[1];
        u32x4 b3l = ((const u32x4*)wr3)[0], b3h = ((const u32x4*)wr3)[1];

#pragma unroll 4
        for (int kc = 0; kc < DD; kc += 32) {
            v16bf af = make_frag(al, ah);
            v16bf f0 = make_frag(b0l, b0h);
            v16bf f1 = make_frag(b1l, b1h);
            v16bf f2 = make_frag(b2l, b2h);
            v16bf f3 = make_frag(b3l, b3h);
            const int kn = kc + 32;
            if (kn < DD) {                             // prefetch next K-step
                al  = *(const u32x4*)(arow + kn);
                ah  = *(const u32x4*)(arow + kn + 8);
                b0l = ((const u32x4*)(wr0 + kn))[0]; b0h = ((const u32x4*)(wr0 + kn))[1];
                b1l = ((const u32x4*)(wr1 + kn))[0]; b1h = ((const u32x4*)(wr1 + kn))[1];
                b2l = ((const u32x4*)(wr2 + kn))[0]; b2h = ((const u32x4*)(wr2 + kn))[1];
                b3l = ((const u32x4*)(wr3 + kn))[0]; b3h = ((const u32x4*)(wr3 + kn))[1];
            }
            c0 = __builtin_amdgcn_wmma_f32_16x16x32_bf16(false, af, false, f0, (short)0, c0, false, false);
            c1 = __builtin_amdgcn_wmma_f32_16x16x32_bf16(false, af, false, f1, (short)0, c1, false, false);
            c2 = __builtin_amdgcn_wmma_f32_16x16x32_bf16(false, af, false, f2, (short)0, c2, false, false);
            c3 = __builtin_amdgcn_wmma_f32_16x16x32_bf16(false, af, false, f3, (short)0, c3, false, false);
        }

        // epilogue: tanh(z + bias[h]) * gate[b,h], accumulate per row
        auto epi = [&](v8f& c, int j) {
            const int   h  = (ntb + j) * 16 + l16;
            const float bi = bias[h];
            const float g  = gate[b * HH + h];
#pragma unroll
            for (int i = 0; i < 8; ++i)
                acc[i] += fast_tanh(c[i] + bi) * g;    // c[i]: row = half*8+i, col = l16
        };
        epi(c0, 0); epi(c1, 1); epi(c2, 2); epi(c3, 3);
    }

    // reduce over the 16 lanes of each half (16 N-columns each)
#pragma unroll
    for (int i = 0; i < 8; ++i) {
        float v = acc[i];
        for (int off = 8; off; off >>= 1) v += __shfl_xor(v, off, 16);
        acc[i] = v;
    }
    if (l16 == 0) {
        float* dst = logits_part + ngroup * MS + mbase + mrow0 + half * 8;
#pragma unroll
        for (int i = 0; i < 8; ++i) dst[i] = acc[i];
    }
}

// ---------------------------------------------------------------------------
// Kernel 3: per-batch softmax over S for both logit sets + product renorm.
// grid = (B), block = S = 512. Sums the two N-half partials, then writes
// final weights [S,B] into the d_out tail.
// ---------------------------------------------------------------------------
__global__ __launch_bounds__(512)
void softmax_combine_kernel(const float* __restrict__ lh, const float* __restrict__ lf,
                            float* __restrict__ wout) {
    __shared__ float red[SS];
    const int b = blockIdx.x;
    const int s = threadIdx.x;
    const int m = b * SS + s;

    float xh = lh[m] + lh[MS + m];
    float xf = lf[m] + lf[MS + m];

    auto blockMax = [&](float v) -> float {
        red[s] = v; __syncthreads();
        for (int o = SS / 2; o > 0; o >>= 1) {
            if (s < o) red[s] = fmaxf(red[s], red[s + o]);
            __syncthreads();
        }
        float r = red[0]; __syncthreads(); return r;
    };
    auto blockSum = [&](float v) -> float {
        red[s] = v; __syncthreads();
        for (int o = SS / 2; o > 0; o >>= 1) {
            if (s < o) red[s] += red[s + o];
            __syncthreads();
        }
        float r = red[0]; __syncthreads(); return r;
    };

    float mh = blockMax(xh);
    float eh = __expf(xh - mh);
    float wh = eh / (EPSV + blockSum(eh));

    float mf = blockMax(xf);
    float ef = __expf(xf - mf);
    float wf = ef / (EPSV + blockSum(ef));

    float p  = wh * wf;
    float w  = p / (EPSV + blockSum(p));

    wout[s * BB + b] = w;                              // weights layout [S, B, 1]
}

// ---------------------------------------------------------------------------
// Kernel 4: context[b,d] = sum_s hs[b,s,d] * w[s,b].  grid = (D/256, B)
// ---------------------------------------------------------------------------
__global__ __launch_bounds__(256)
void context_kernel(const float* __restrict__ hs, const float* __restrict__ w,
                    float* __restrict__ ctx) {
    __shared__ float ws[SS];
    const int b = blockIdx.y;
    const int d = blockIdx.x * 256 + threadIdx.x;
    for (int i = threadIdx.x; i < SS; i += 256) ws[i] = w[i * BB + b];
    __syncthreads();
    float a = 0.f;
#pragma unroll 4
    for (int s = 0; s < SS; ++s)
        a += hs[((size_t)b * SS + s) * DD + d] * ws[s];
    ctx[b * DD + d] = a;
}

// ---------------------------------------------------------------------------
// Kernel 5: out[b,h] = finished[b] ? 0 : [ctx | input] . W3[h] + b3[h]
// grid = (H/256, B)
// ---------------------------------------------------------------------------
__global__ __launch_bounds__(256)
void out_kernel(const float* __restrict__ ctx, const float* __restrict__ inp,
                const float* __restrict__ W3, const float* __restrict__ b3,
                const unsigned char* __restrict__ finished, float* __restrict__ out) {
    __shared__ float xc[DD];
    __shared__ float xi[DD];
    const int b = blockIdx.y;
    const int h = blockIdx.x * 256 + threadIdx.x;
    for (int i = threadIdx.x; i < DD; i += 256) {
        xc[i] = ctx[b * DD + i];
        xi[i] = inp[b * DD + i];
    }
    __syncthreads();
    float acc = b3[h];
    const float4* w0 = (const float4*)(W3 + (size_t)h * (2 * DD));
#pragma unroll 4
    for (int k4 = 0; k4 < DD / 4; ++k4) {
        float4 w = w0[k4];
        const float* xp = &xc[k4 * 4];
        acc += w.x * xp[0] + w.y * xp[1] + w.z * xp[2] + w.w * xp[3];
    }
    const float4* w1 = w0 + DD / 4;
#pragma unroll 4
    for (int k4 = 0; k4 < DD / 4; ++k4) {
        float4 w = w1[k4];
        const float* xp = &xi[k4 * 4];
        acc += w.x * xp[0] + w.y * xp[1] + w.z * xp[2] + w.w * xp[3];
    }
    out[b * HH + h] = finished[b] ? 0.f : acc;
}

// ---------------------------------------------------------------------------
// Host launcher
// ---------------------------------------------------------------------------
extern "C" void kernel_launch(void* const* d_in, const int* in_sizes, int n_in,
                              void* d_out, int out_size, void* d_ws, size_t ws_size,
                              hipStream_t stream) {
    // inputs (setup_inputs order)
    const float* hidden = (const float*)d_in[0];    // [B,S,D]
    const float* field  = (const float*)d_in[1];    // [B,S,F]
    const float* inp    = (const float*)d_in[2];    // [B,D]
    const unsigned char* finished = (const unsigned char*)d_in[3]; // [B,1] bool
    const float* W1 = (const float*)d_in[4];  const float* b1 = (const float*)d_in[5];
    const float* W2 = (const float*)d_in[6];  const float* b2 = (const float*)d_in[7];
    const float* W3 = (const float*)d_in[8];  const float* b3 = (const float*)d_in[9];
    const float* W4 = (const float*)d_in[10]; const float* b4 = (const float*)d_in[11];
    const float* W5 = (const float*)d_in[12]; const float* b5 = (const float*)d_in[13];

    float* out_main = (float*)d_out;                 // [B,H]
    float* out_w    = (float*)d_out + BB * HH;       // [S,B,1]

    // workspace layout (bytes)
    char* ws = (char*)d_ws;
    float*          gamma = (float*)(ws + 0);                           // 256 KB
    float*          alpha = (float*)(ws + 262144);                      // 256 KB
    float*          lg_h  = (float*)(ws + 524288);                      // 2 x 128 KB partials
    float*          lg_f  = (float*)(ws + 786432);                      // 2 x 128 KB partials
    float*          ctx   = (float*)(ws + 1048576);                     // 256 KB
    unsigned short* W1b   = (unsigned short*)(ws + 1310720);            // 2 MB
    unsigned short* W4b   = (unsigned short*)(ws + 3407872);            // 2 MB
    (void)in_sizes; (void)n_in; (void)out_size; (void)ws_size;

    // 0) weights -> bf16
    convert_w_kernel<<<(HH * DD) / 256, 256, 0, stream>>>(W1, W4, W1b, W4b);

    // 1) gates
    gate_kernel<<<dim3(HH / 256, BB, 2), 256, 0, stream>>>(inp, W2, b2, W5, b5, gamma, alpha);

    // 2) big fused WMMA logits (hidden/W1/gamma and field/W4/alpha)
    const size_t lds_bytes = (size_t)M_TILE * PITCH * sizeof(unsigned short); // ~129 KB
    attn_logits_kernel<<<MS / M_TILE, 256, lds_bytes, stream>>>(hidden, W1b, b1, gamma, lg_h);
    attn_logits_kernel<<<MS / M_TILE, 256, lds_bytes, stream>>>(field,  W4b, b4, alpha, lg_f);

    // 3) softmaxes + product renorm -> final weights (directly into d_out tail)
    softmax_combine_kernel<<<BB, SS, 0, stream>>>(lg_h, lg_f, out_w);

    // 4) context = sum_s hs * w
    context_kernel<<<dim3(DD / 256, BB), 256, 0, stream>>>(hidden, out_w, ctx);

    // 5) final projection + finished mask
    out_kernel<<<dim3(HH / 256, BB), 256, 0, stream>>>(ctx, inp, W3, b3, finished, out_main);
}
